// Policy_GNN_63574105915525
// MI455X (gfx1250) — matrix operand order repaired and checked by reading, not compiled, not run on hardware
//
#include <hip/hip_runtime.h>
#include <hip/hip_bf16.h>

typedef __attribute__((ext_vector_type(16))) _Float16 v16h;
typedef __attribute__((ext_vector_type(8)))  float    v8f;

#define HDIM 128
#define EPSV 1e-8f
#define ALPHAV 0.3f

// ---------------------------------------------------------------------------
// Weight prep: Wt16[k*128 + n] = (f16) W[n*128 + k]   (transposed f16 copy)
// ---------------------------------------------------------------------------
__global__ void k_w2h(const float* __restrict__ W, _Float16* __restrict__ out)
{
    int idx = blockIdx.x * blockDim.x + threadIdx.x;
    if (idx < 128 * 128) {
        int k = idx >> 7, n = idx & 127;
        out[idx] = (_Float16)W[n * 128 + k];
    }
}

// ---------------------------------------------------------------------------
// WMMA GEMM: Y[r,0:128] = X[r,0:128] @ W^T (+bias)(+relu)
// Wt16 is the pre-transposed f16 weight (row k holds W^T[k, 0:128]).
// block = 256 (8 waves); each wave computes a 16x128 output strip.
// Weight tile staged into LDS with async global->LDS copies (ASYNCcnt path).
// ---------------------------------------------------------------------------
__global__ void __launch_bounds__(256)
k_gemm128(const float* __restrict__ X, const _Float16* __restrict__ Wt16,
          const float* __restrict__ bias, float* __restrict__ Y,
          int rows, int do_bias, int do_relu)
{
    __shared__ __align__(32) _Float16 Wt[128][144]; // +16 halfs pad: 32B-aligned rows

    // --- async stage: 128 rows x 256B from global into padded LDS rows ---
    {
        const unsigned ldsbase = (unsigned)(size_t)(&Wt[0][0]); // low32 of flat = LDS offset
        const char* gbase = (const char*)Wt16;
#pragma unroll
        for (int i = 0; i < 8; ++i) {
            int c = threadIdx.x + i * 256;      // 2048 16B chunks
            int row = c >> 4;
            int off = (c & 15) << 4;
            unsigned la = ldsbase + row * 288 + off;
            const char* ga = gbase + row * 256 + off;
            asm volatile("global_load_async_to_lds_b128 %0, %1, off"
                         :: "v"(la), "v"(ga) : "memory");
        }
        asm volatile("s_wait_asynccnt 0" ::: "memory");
    }
    __syncthreads();

    const int wave = threadIdx.x >> 5;
    const int lane = threadIdx.x & 31;
    const int m0 = (blockIdx.x * 8 + wave) * 16;
    if (m0 >= rows) return;

    const int rA = m0 + (lane & 15);
    const int rAc = rA < rows ? rA : rows - 1;      // clamp; stores are guarded
    const int kbase = (lane < 16) ? 0 : 8;
    const float* __restrict__ xrow = X + (size_t)rAc * HDIM;
    __builtin_prefetch(xrow, 0, 2);

    // --- preload ALL A data (4 k-steps) to keep loads pipelined ---
    float4 fa[16];
#pragma unroll
    for (int t = 0; t < 4; ++t) {
        fa[4*t+0] = *(const float4*)(xrow + 32*t + kbase);
        fa[4*t+1] = *(const float4*)(xrow + 32*t + kbase + 4);
        fa[4*t+2] = *(const float4*)(xrow + 32*t + 16 + kbase);
        fa[4*t+3] = *(const float4*)(xrow + 32*t + 16 + kbase + 4);
    }
#define CVT8(dst, o, A, B) \
    dst[o+0]=(_Float16)(A).x; dst[o+1]=(_Float16)(A).y; dst[o+2]=(_Float16)(A).z; dst[o+3]=(_Float16)(A).w; \
    dst[o+4]=(_Float16)(B).x; dst[o+5]=(_Float16)(B).y; dst[o+6]=(_Float16)(B).z; dst[o+7]=(_Float16)(B).w;
    v16h afr[4];
#pragma unroll
    for (int t = 0; t < 4; ++t) {
        CVT8(afr[t], 0, fa[4*t+0], fa[4*t+1]);
        CVT8(afr[t], 8, fa[4*t+2], fa[4*t+3]);
    }
#undef CVT8

    v8f acc[8];
#pragma unroll
    for (int nt = 0; nt < 8; ++nt) acc[nt] = (v8f){0.f,0.f,0.f,0.f,0.f,0.f,0.f,0.f};

#pragma unroll
    for (int t = 0; t < 4; ++t) {
        const int k0 = t * 32;
#pragma unroll
        for (int nt = 0; nt < 8; ++nt) {
            v16h b;
#pragma unroll
            for (int s = 0; s < 16; ++s)
                b[s] = Wt[k0 + kbase + (s < 8 ? s : s + 8)][nt * 16 + (lane & 15)];
            acc[nt] = __builtin_amdgcn_wmma_f32_16x16x32_f16(
                false, afr[t], false, b, (short)0, acc[nt], false, false);
        }
    }

    // C/D layout: vgpr r -> M = m0 + r + (lane>=16 ? 8 : 0), N = lane&15
    const int rowoff = (lane >= 16) ? 8 : 0;
    const int nlo = lane & 15;
#pragma unroll
    for (int nt = 0; nt < 8; ++nt) {
        const int col = nt * 16 + nlo;
        const float bv = do_bias ? bias[col] : 0.f;
#pragma unroll
        for (int r = 0; r < 8; ++r) {
            const int row = m0 + r + rowoff;
            if (row < rows) {
                float v = acc[nt][r] + bv;
                if (do_relu) v = fmaxf(v, 0.f);
                Y[(size_t)row * HDIM + col] = v;
            }
        }
    }
}

// ---------------------------------------------------------------------------
// graph prep
// ---------------------------------------------------------------------------
__global__ void k_deg_count(const int* __restrict__ dst, float* __restrict__ deg, int E)
{
    int e = blockIdx.x * blockDim.x + threadIdx.x;
    if (e < E) atomicAdd(&deg[dst[e]], 1.0f);
}

__global__ void k_deg_fin(float* __restrict__ inv, float* __restrict__ dis, int n)
{
    int i = blockIdx.x * blockDim.x + threadIdx.x;
    if (i < n) {
        float d = inv[i] + 1.0f;   // self loop
        dis[i] = rsqrtf(d);
        inv[i] = 1.0f / d;
    }
}

__global__ void k_norm(float* __restrict__ ne, const float* __restrict__ dis,
                       const int* __restrict__ src, const int* __restrict__ dst, int E)
{
    int e = blockIdx.x * blockDim.x + threadIdx.x;
    if (e < E) ne[e] = dis[src[e]] * dis[dst[e]];
}

// ---------------------------------------------------------------------------
// GCN pieces
// ---------------------------------------------------------------------------
__global__ void k_selfinit(float* __restrict__ out, const float* __restrict__ xw,
                           const float* __restrict__ inv, const float* __restrict__ b,
                           int nh)
{
    int idx = blockIdx.x * blockDim.x + threadIdx.x;
    if (idx < nh) {
        int i = idx >> 7, j = idx & 127;
        out[idx] = xw[idx] * inv[i] + b[j];
    }
}

__global__ void k_scatter(float* __restrict__ out, const float* __restrict__ xw,
                          const float* __restrict__ ne,
                          const int* __restrict__ src, const int* __restrict__ dst, int E)
{
    long long gid = (long long)blockIdx.x * blockDim.x + threadIdx.x;
    int e = (int)(gid >> 5);
    if (e >= E) return;
    int j = ((int)gid & 31) * 4;
    int s = src[e], d = dst[e];
    float w = ne[e];
    const float4 v = *(const float4*)(xw + (size_t)s * HDIM + j);
    float* p = out + (size_t)d * HDIM + j;
    atomicAdd(p + 0, v.x * w);
    atomicAdd(p + 1, v.y * w);
    atomicAdd(p + 2, v.z * w);
    atomicAdd(p + 3, v.w * w);
}

// one-hot query GCN init: out[i,j] = b[j] + (i==q ? w[j]*inv[q] : 0)
__global__ void k_qinit(float* __restrict__ out, const float* __restrict__ wcol,
                        const float* __restrict__ b, const float* __restrict__ inv,
                        const int* __restrict__ qp, int nh)
{
    int idx = blockIdx.x * blockDim.x + threadIdx.x;
    if (idx < nh) {
        int i = idx >> 7, j = idx & 127;
        int q = qp[0];
        float v = b[j];
        if (i == q) v += wcol[j] * inv[q];
        out[idx] = v;
    }
}

__global__ void k_qscatter(float* __restrict__ out, const float* __restrict__ wcol,
                           const float* __restrict__ ne,
                           const int* __restrict__ src, const int* __restrict__ dst,
                           const int* __restrict__ qp, int E)
{
    int e = blockIdx.x * blockDim.x + threadIdx.x;
    if (e >= E) return;
    if (src[e] != qp[0]) return;
    float w = ne[e];
    float* p = out + (size_t)dst[e] * HDIM;
    for (int j = 0; j < HDIM; ++j) atomicAdd(p + j, wcol[j] * w);
}

// buf[i,j] += b_lq[j] + (i==q ? w_lq[j] : 0)   (querys_ + feats_)
__global__ void k_addquery(float* __restrict__ buf, const float* __restrict__ blq,
                           const float* __restrict__ wlq, const int* __restrict__ qp, int nh)
{
    int idx = blockIdx.x * blockDim.x + threadIdx.x;
    if (idx < nh) {
        int i = idx >> 7, j = idx & 127;
        float v = buf[idx] + blq[j];
        if (i == qp[0]) v += wlq[j];
        buf[idx] = v;
    }
}

__global__ void k_relu_ip(float* __restrict__ buf, int n)
{
    int i = blockIdx.x * blockDim.x + threadIdx.x;
    if (i < n) buf[i] = fmaxf(buf[i], 0.f);
}

__global__ void k_copy_opt_relu(float* __restrict__ d, const float* __restrict__ s,
                                int n, int relu)
{
    int i = blockIdx.x * blockDim.x + threadIdx.x;
    if (i < n) {
        float v = s[i];
        d[i] = relu ? fmaxf(v, 0.f) : v;
    }
}

__global__ void k_add3(float* __restrict__ d, const float* __restrict__ a,
                       const float* __restrict__ b, const float* __restrict__ c,
                       int n, int relu)
{
    int i = blockIdx.x * blockDim.x + threadIdx.x;
    if (i < n) {
        float v = a[i] + b[i] + c[i];
        d[i] = relu ? fmaxf(v, 0.f) : v;
    }
}

// ---------------------------------------------------------------------------
// decode
// ---------------------------------------------------------------------------
__global__ void k_qemb(const float* __restrict__ x, const int* __restrict__ qp,
                       float* __restrict__ qemb, float* __restrict__ misc)
{
    __shared__ float red[128];
    int q = qp[0];
    int j = threadIdx.x;
    float v = x[(size_t)q * HDIM + j];
    qemb[j] = v;
    red[j] = v * v;
    __syncthreads();
    for (int s = 64; s; s >>= 1) {
        if (j < s) red[j] += red[j + s];
        __syncthreads();
    }
    if (j == 0) misc[2] = fmaxf(sqrtf(red[0]), EPSV);
}

// one wave per node: cosine(x_i, q_emb) + mlp2 head + sigmoid
__global__ void __launch_bounds__(256)
k_scores(const float* __restrict__ x, const float* __restrict__ qemb,
         const float* __restrict__ misc,
         const float* __restrict__ m2w1, const float* __restrict__ m2b1,
         const float* __restrict__ m2w2, const float* __restrict__ m2b2,
         float* __restrict__ out, int n)
{
    int wave = threadIdx.x >> 5, lane = threadIdx.x & 31;
    int node = blockIdx.x * 8 + wave;
    if (node >= n) return;
    const float4 v  = *(const float4*)(x + (size_t)node * HDIM + lane * 4);
    const float4 qe = *(const float4*)(qemb + lane * 4);
    const float4 wv = *(const float4*)(m2w1 + lane * 4);
    float dq = v.x*qe.x + v.y*qe.y + v.z*qe.z + v.w*qe.w;
    float dx = v.x*v.x + v.y*v.y + v.z*v.z + v.w*v.w;
    float dw = v.x*wv.x + v.y*wv.y + v.z*wv.z + v.w*wv.w;
    for (int off = 16; off; off >>= 1) {
        dq += __shfl_xor(dq, off, 32);
        dx += __shfl_xor(dx, off, 32);
        dw += __shfl_xor(dw, off, 32);
    }
    if (lane == 0) {
        float qn = misc[2];
        float vn = fmaxf(sqrtf(dx), EPSV);
        float sim_xq = dq / (vn * qn);
        float sim_xx = m2w2[0] * fmaxf(dw + m2b1[0], 0.f) + m2b2[0];
        float sim = ((1.f - ALPHAV) * sim_xq + ALPHAV * sim_xx) * 2.f;
        out[node] = 1.f / (1.f + expf(-sim));
    }
}

__device__ __forceinline__ float splus(float t)
{
    return log1pf(expf(-fabsf(t))) + fmaxf(t, 0.f);
}

// one wave per edge: softplus(sign * cos((x_a+x_b)/2, q_emb)) accumulated
__global__ void __launch_bounds__(256)
k_loss(const float* __restrict__ x, const int* __restrict__ ia, const int* __restrict__ ib,
       const float* __restrict__ qemb, const float* __restrict__ misc,
       float* __restrict__ acc, int E, int negate)
{
    int wave = threadIdx.x >> 5, lane = threadIdx.x & 31;
    int e = blockIdx.x * 8 + wave;
    if (e >= E) return;
    int a = ia[e], b = ib[e];
    const float4 va = *(const float4*)(x + (size_t)a * HDIM + lane * 4);
    const float4 vb = *(const float4*)(x + (size_t)b * HDIM + lane * 4);
    const float4 qe = *(const float4*)(qemb + lane * 4);
    float mx = (va.x + vb.x) * 0.5f, my = (va.y + vb.y) * 0.5f;
    float mz = (va.z + vb.z) * 0.5f, mw = (va.w + vb.w) * 0.5f;
    float dq = mx*qe.x + my*qe.y + mz*qe.z + mw*qe.w;
    float dv = mx*mx + my*my + mz*mz + mw*mw;
    for (int off = 16; off; off >>= 1) {
        dq += __shfl_xor(dq, off, 32);
        dv += __shfl_xor(dv, off, 32);
    }
    if (lane == 0) {
        float qn = misc[2];
        float s = dq / (fmaxf(sqrtf(dv), EPSV) * qn);
        float t = negate ? s : -s;          // pos edges: softplus(-s); neg: softplus(s)
        atomicAdd(acc, splus(t));
    }
}

__global__ void k_loss_fin(float* __restrict__ out, const float* __restrict__ misc, int E)
{
    if (threadIdx.x == 0 && blockIdx.x == 0)
        out[0] = misc[0] / (float)E + misc[1] / (float)E;
}

// ---------------------------------------------------------------------------
// host driver
// ---------------------------------------------------------------------------
extern "C" void kernel_launch(void* const* d_in, const int* in_sizes, int n_in,
                              void* d_out, int out_size, void* d_ws, size_t ws_size,
                              hipStream_t stream)
{
    const int N = in_sizes[1] / HDIM;      // 50000
    const int E = in_sizes[2] / 2;         // 800000
    const int NH = N * HDIM;

    const int*   qp    = (const int*)d_in[0];
    const float* feats = (const float*)d_in[1];
    const int*   esrc  = (const int*)d_in[2];
    const int*   edst  = esrc + E;
    const int*   nsrc  = (const int*)d_in[3];
    const int*   ndst  = nsrc + E;
    const float* w_lq = (const float*)d_in[4];   const float* b_lq = (const float*)d_in[5];
    const float* w_lf = (const float*)d_in[6];   const float* b_lf = (const float*)d_in[7];
    const float* wq0  = (const float*)d_in[8];   const float* bq0  = (const float*)d_in[9];
    const float* wq1  = (const float*)d_in[10];  const float* bq1  = (const float*)d_in[11];
    const float* wq2  = (const float*)d_in[12];  const float* bq2  = (const float*)d_in[13];
    const float* wx0  = (const float*)d_in[14];  const float* bx0  = (const float*)d_in[15];
    const float* wx1  = (const float*)d_in[16];  const float* bx1  = (const float*)d_in[17];
    const float* wx2  = (const float*)d_in[18];  const float* bx2  = (const float*)d_in[19];
    const float* wf0  = (const float*)d_in[20];  const float* bf0  = (const float*)d_in[21];
    const float* wf1  = (const float*)d_in[22];  const float* bf1  = (const float*)d_in[23];
    const float* wf2  = (const float*)d_in[24];  const float* bf2  = (const float*)d_in[25];
    const float* m1w1 = (const float*)d_in[26];  const float* m1b1 = (const float*)d_in[27];
    const float* m1w2 = (const float*)d_in[28];  const float* m1b2 = (const float*)d_in[29];
    const float* m2w1 = (const float*)d_in[30];  const float* m2b1 = (const float*)d_in[31];
    const float* m2w2 = (const float*)d_in[32];  const float* m2b2 = (const float*)d_in[33];
    float* out = (float*)d_out;

    // workspace layout (floats)
    float* w = (float*)d_ws;
    float* s_inv  = w;                       // N   (also degree accumulator)
    float* s_dis  = w + N;                   // N
    float* s_norm = w + 2 * N;               // E
    float* s_misc = w + 2 * N + E;           // 256: [0]=posAcc [1]=negAcc [2]=qn [32..]=qemb
    float* s_qemb = s_misc + 32;
    _Float16* s_wt = (_Float16*)(w + 2 * N + E + 256);   // 11 x 128x128 f16 (transposed)
    float* big    = w + 2 * N + E + 256 + (11 * 16384) / 2;
    float* b_xw  = big;
    float* b_agg = big + (size_t)NH;         // doubles as final x
    float* b_hq  = big + (size_t)2 * NH;
    float* b_h   = big + (size_t)3 * NH;
    float* b_hf  = big + (size_t)4 * NH;
    float* b_qf  = big + (size_t)5 * NH;

    // weight slots: 0 w_lf, 1 wx0, 2 wf0, 3 wq1, 4 wx1, 5 wf1,
    //               6 wq2, 7 wx2, 8 wf2, 9 m1w1, 10 m1w2
    const float* wsrc[11] = {w_lf, wx0, wf0, wq1, wx1, wf1, wq2, wx2, wf2, m1w1, m1w2};
    enum { S_WLF=0, S_WX0, S_WF0, S_WQ1, S_WX1, S_WF1, S_WQ2, S_WX2, S_WF2, S_M1W1, S_M1W2 };

    const int B = 256;
    const int gNH = (NH + B - 1) / B;
    const int gN  = (N + B - 1) / B;
    const int gE  = (E + B - 1) / B;
    const int gE32 = (int)(((long long)E * 32 + B - 1) / B);
    const int gGemm = ((N + 15) / 16 + 7) / 8;
    const int gWave8N = (N + 7) / 8;
    const int gWave8E = (E + 7) / 8;

    // ---- convert weights once (f32 -> transposed f16) ----
    for (int s = 0; s < 11; ++s)
        k_w2h<<<64, 256, 0, stream>>>(wsrc[s], s_wt + (size_t)s * 16384);

    auto gemm = [&](const float* X, int wslot, const float* bias, float* Y,
                    int bias_f, int relu_f) {
        k_gemm128<<<gGemm, 256, 0, stream>>>(X, s_wt + (size_t)wslot * 16384,
                                             bias, Y, N, bias_f, relu_f);
    };
    auto gcn_into = [&](const float* X, int wslot, const float* bias, float* O) {
        gemm(X, wslot, nullptr, b_xw, 0, 0);
        k_selfinit<<<gNH, B, 0, stream>>>(O, b_xw, s_inv, bias, NH);
        k_scatter<<<gE32, B, 0, stream>>>(O, b_xw, s_norm, esrc, edst, E);
    };

    // ---- graph normalization ----
    hipMemsetAsync(s_inv, 0, (size_t)N * sizeof(float), stream);
    hipMemsetAsync(s_misc, 0, 256 * sizeof(float), stream);
    k_deg_count<<<gE, B, 0, stream>>>(edst, s_inv, E);
    k_deg_fin<<<gN, B, 0, stream>>>(s_inv, s_dis, N);
    k_norm<<<gE, B, 0, stream>>>(s_norm, s_dis, esrc, edst, E);

    // ---- layer 0 ----
    k_qinit<<<gNH, B, 0, stream>>>(b_hq, wq0, bq0, s_inv, qp, NH);
    k_qscatter<<<gE, B, 0, stream>>>(b_hq, wq0, s_norm, esrc, edst, qp, E);
    k_relu_ip<<<gNH, B, 0, stream>>>(b_hq, NH);
    gcn_into(feats, S_WX0, bx0, b_h);
    k_relu_ip<<<gNH, B, 0, stream>>>(b_h, NH);
    gemm(feats, S_WLF, b_lf, b_qf, 1, 0);
    k_addquery<<<gNH, B, 0, stream>>>(b_qf, b_lq, w_lq, qp, NH);
    gcn_into(b_qf, S_WF0, bf0, b_agg);
    k_add3<<<gNH, B, 0, stream>>>(b_hf, b_hq, b_h, b_agg, NH, 1);

    // ---- layer 1 ----
    gcn_into(b_hq, S_WQ1, bq1, b_agg);
    k_copy_opt_relu<<<gNH, B, 0, stream>>>(b_hq, b_agg, NH, 1);
    gcn_into(b_h, S_WX1, bx1, b_agg);
    k_copy_opt_relu<<<gNH, B, 0, stream>>>(b_h, b_agg, NH, 1);
    gcn_into(b_hf, S_WF1, bf1, b_agg);
    k_add3<<<gNH, B, 0, stream>>>(b_hf, b_hq, b_h, b_agg, NH, 1);

    // ---- layer 2 (no relu) ----
    gcn_into(b_hq, S_WQ2, bq2, b_agg);
    k_copy_opt_relu<<<gNH, B, 0, stream>>>(b_hq, b_agg, NH, 0);
    gcn_into(b_h, S_WX2, bx2, b_agg);
    k_copy_opt_relu<<<gNH, B, 0, stream>>>(b_h, b_agg, NH, 0);
    gcn_into(b_hf, S_WF2, bf2, b_agg);
    k_add3<<<gNH, B, 0, stream>>>(b_hf, b_hq, b_h, b_agg, NH, 0);

    // ---- mlp1: x = lin(relu(lin(hf, m1_w1, m1_b1)), m1_w2, m1_b2) ----
    gemm(b_hf, S_M1W1, m1b1, b_xw, 1, 1);
    gemm(b_xw, S_M1W2, m1b2, b_agg, 1, 0);   // x lives in b_agg

    // ---- decode ----
    k_qemb<<<1, 128, 0, stream>>>(b_agg, qp, s_qemb, s_misc);
    k_scores<<<gWave8N, 256, 0, stream>>>(b_agg, s_qemb, s_misc,
                                          m2w1, m2b1, m2w2, m2b2, out, N);
    k_loss<<<gWave8E, 256, 0, stream>>>(b_agg, esrc, edst, s_qemb, s_misc,
                                        s_misc + 0, E, 0);
    k_loss<<<gWave8E, 256, 0, stream>>>(b_agg, nsrc, ndst, s_qemb, s_misc,
                                        s_misc + 1, E, 1);
    k_loss_fin<<<1, 32, 0, stream>>>(out + N, s_misc, E);
}
